// ProcessSingleEvent_26328149524729
// MI455X (gfx1250) — 8-bit matrix operand order repaired and checked by reading, compiled, not run on hardware
//
#include <hip/hip_runtime.h>
#include <hip/hip_bf16.h>

typedef int v8i __attribute__((ext_vector_type(8)));

#define N_HITS 8192
#define E_MAX  8192
#define SENT_KEY (1 << 20)

__global__ void hg_normalize(const float* __restrict__ ed,
                             float* __restrict__ X,
                             float* __restrict__ phin, float* __restrict__ zn,
                             int* __restrict__ ev, int* __restrict__ st,
                             int* __restrict__ trk) {
  int i = blockIdx.x * blockDim.x + threadIdx.x;
  if (i >= N_HITS) return;
  float x = ed[i * 6 + 1];
  float y = ed[i * 6 + 2];
  float z = ed[i * 6 + 3];
  float r   = sqrtf(x * x + y * y);
  float phi = atan2f(x, y);
  float rn = 2.0f * (r - 269.0f) / (581.0f - 269.0f) - 1.0f;
  float pn = 2.0f * (phi + 3.15f) / (2.0f * 3.15f) - 1.0f;
  float zz = 2.0f * (z + 2386.0f) / (2.0f * 2386.0f) - 1.0f;
  X[i * 3 + 0] = rn;
  X[i * 3 + 1] = pn;
  X[i * 3 + 2] = zz;
  phin[i] = pn;
  zn[i]   = zz;
  ev[i]   = (int)ed[i * 6 + 0];
  st[i]   = (int)ed[i * 6 + 4];
  trk[i]  = (int)ed[i * 6 + 5];
}

__device__ __forceinline__ bool pair_valid(int evi, int sti, float pi, float zi,
                                           int evj, int stj, float pj, float zj) {
  float dphi = pj - pi;
  float dz   = zj - zi;
  return (evi == evj) & (sti == stj - 1) &
         (dphi > -0.04f) & (dphi < 0.04f) &
         (dz   > -0.03f) & (dz   < 0.03f);
}

__global__ void hg_rowcount(const float* __restrict__ phin, const float* __restrict__ zn,
                            const int* __restrict__ ev, const int* __restrict__ st,
                            int* __restrict__ rcnt) {
  int lane = threadIdx.x & 31;
  int row  = blockIdx.x * (blockDim.x >> 5) + (threadIdx.x >> 5);
  int   evi = ev[row], sti = st[row];
  float pi  = phin[row], zi = zn[row];
  int cnt = 0;
  for (int j0 = 0; j0 < N_HITS; j0 += 32) {
    int j = j0 + lane;
    bool v = pair_valid(evi, sti, pi, zi, ev[j], st[j], phin[j], zn[j]);
    unsigned m = __builtin_amdgcn_ballot_w32(v);
    cnt += __builtin_popcount(m);
  }
  if (lane == 0) rcnt[row] = cnt;
}

__global__ void hg_scan(const int* __restrict__ rcnt, int* __restrict__ rbase,
                        int* __restrict__ countp) {
  __shared__ int part[256];
  int t = threadIdx.x;
  int s = 0;
  for (int k = 0; k < 32; ++k) s += rcnt[t * 32 + k];
  part[t] = s;
  __syncthreads();
  if (t == 0) {
    int run = 0;
    for (int i = 0; i < 256; ++i) { int v = part[i]; part[i] = run; run += v; }
    *countp = run;
  }
  __syncthreads();
  int base = part[t];
  for (int k = 0; k < 32; ++k) { rbase[t * 32 + k] = base; base += rcnt[t * 32 + k]; }
}

__global__ void hg_emit(const float* __restrict__ phin, const float* __restrict__ zn,
                        const int* __restrict__ ev, const int* __restrict__ st,
                        const int* __restrict__ trk, const int* __restrict__ rbase,
                        int* __restrict__ ksrc, int* __restrict__ kdst) {
  int lane = threadIdx.x & 31;
  int row  = blockIdx.x * (blockDim.x >> 5) + (threadIdx.x >> 5);
  int   evi = ev[row], sti = st[row], ti = trk[row];
  float pi  = phin[row], zi = zn[row];
  int base = rbase[row];
  for (int j0 = 0; j0 < N_HITS; j0 += 32) {
    int j = j0 + lane;
    bool v = pair_valid(evi, sti, pi, zi, ev[j], st[j], phin[j], zn[j]);
    unsigned m = __builtin_amdgcn_ballot_w32(v);
    int pos = base + __builtin_popcount(m & ((1u << lane) - 1u));
    if (v && pos < E_MAX) {
      ksrc[pos] = ti;
      kdst[pos] = trk[j];
    }
    base += __builtin_popcount(m);
  }
}

__global__ void hg_finalize(const int* __restrict__ countp, const int* __restrict__ trk,
                            int* __restrict__ ksrc, int* __restrict__ kdst,
                            float* __restrict__ Edg, float* __restrict__ Y,
                            float* __restrict__ Msk) {
  int e = blockIdx.x * blockDim.x + threadIdx.x;
  if (e >= E_MAX) return;
  int cnt = *countp;
  bool v = e < cnt;
  float t0 = (float)trk[0];
  Edg[e * 2 + 0] = v ? (float)ksrc[e] : t0;
  Edg[e * 2 + 1] = v ? (float)kdst[e] : t0;
  Y[e]   = 0.0f;
  Msk[e] = v ? 1.0f : 0.0f;
  if (!v) { ksrc[e] = SENT_KEY; kdst[e] = SENT_KEY; }
}

__device__ __forceinline__ unsigned oh4(int x, int c) {
  unsigned t = (unsigned)(x - c);
  return (t < 4u) ? (1u << (t * 8)) : 0u;
}

__global__ void hg_rio_wmma(const int* __restrict__ trk,
                            const int* __restrict__ ksrc, const int* __restrict__ kdst,
                            float* __restrict__ Ri, float* __restrict__ Ro) {
  int lane = threadIdx.x & 31;
  int wave = (blockIdx.x * blockDim.x + threadIdx.x) >> 5;
  int tile_i = wave >> 9;
  int tile_e = wave & 511;
  int m    = lane & 15;
  int half = lane >> 4;

  int ti  = trk[tile_i * 16 + m];
  int hiA = ti >> 5, loA = ti & 31;
  int kd  = kdst[tile_e * 16 + m];
  int hiD = kd >> 5, loD = kd & 31;
  int ks  = ksrc[tile_e * 16 + m];
  int hiS = ks >> 5, loS = ks & 31;

  v8i a, bd, bs;
#pragma unroll
  for (int v = 0; v < 8; ++v) {
    int p = v >> 1, low = v & 1;
    int c = (v >> 2) * 32 + half * 16 + (v & 3) * 4;
    a[v] = (int)((p < 2) ? oh4(hiA, c) : oh4(loA, c - 32));
  }
#pragma unroll
  for (int v = 0; v < 8; ++v) {
    int g = v >> 2, w = v & 3;
    int c = g * 32 + half * 16 + w * 4;
    bd[v] = (int)((g == 0) ? oh4(hiD, c) : oh4(loD, c - 32));
    bs[v] = (int)((g == 0) ? oh4(hiS, c) : oh4(loS, c - 32));
  }

  v8i cz = {};
  v8i d_ri = __builtin_amdgcn_wmma_i32_16x16x64_iu8(false, a, false, bd, cz, false, false);
  v8i d_ro = __builtin_amdgcn_wmma_i32_16x16x64_iu8(false, a, false, bs, cz, false, false);

  size_t col = (size_t)(tile_e * 16 + m);
#pragma unroll
  for (int r = 0; r < 8; ++r) {
    size_t row = (size_t)(tile_i * 16 + half * 8 + r);
    size_t idx = row * (size_t)E_MAX + col;
    Ri[idx] = (d_ri[r] == 2) ? 1.0f : 0.0f;
    Ro[idx] = (d_ro[r] == 2) ? 1.0f : 0.0f;
  }
}

extern "C" void kernel_launch(void* const* d_in, const int* in_sizes, int n_in,
                              void* d_out, int out_size, void* d_ws, size_t ws_size,
                              hipStream_t stream) {
  const float* ed = (const float*)d_in[0];

  float* out = (float*)d_out;
  float* X   = out;
  float* Ri  = X  + (size_t)N_HITS * 3;
  float* Ro  = Ri + (size_t)N_HITS * E_MAX;
  float* Y   = Ro + (size_t)N_HITS * E_MAX;
  float* Edg = Y  + E_MAX;
  float* Msk = Edg + (size_t)E_MAX * 2;

  int*   w      = (int*)d_ws;
  float* phin   = (float*)(w + 0 * N_HITS);
  float* zn     = (float*)(w + 1 * N_HITS);
  int*   ev     = w + 2 * N_HITS;
  int*   st     = w + 3 * N_HITS;
  int*   trk    = w + 4 * N_HITS;
  int*   rcnt   = w + 5 * N_HITS;
  int*   rbase  = w + 6 * N_HITS;
  int*   ksrc   = w + 7 * N_HITS;
  int*   kdst   = w + 8 * N_HITS;
  int*   countp = w + 9 * N_HITS;

  hg_normalize<<<N_HITS / 256, 256, 0, stream>>>(ed, X, phin, zn, ev, st, trk);
  hg_rowcount<<<N_HITS / 8, 256, 0, stream>>>(phin, zn, ev, st, rcnt);
  hg_scan<<<1, 256, 0, stream>>>(rcnt, rbase, countp);
  hg_emit<<<N_HITS / 8, 256, 0, stream>>>(phin, zn, ev, st, trk, rbase, ksrc, kdst);
  hg_finalize<<<E_MAX / 256, 256, 0, stream>>>(countp, trk, ksrc, kdst, Edg, Y, Msk);

  hg_rio_wmma<<<(512 * 512) / 8, 256, 0, stream>>>(trk, ksrc, kdst, Ri, Ro);

  (void)in_sizes; (void)n_in; (void)out_size; (void)ws_size;
}
